// SimpleGraphHead_21835613733002
// MI455X (gfx1250) — matrix operand order repaired and checked by reading, not compiled, not run on hardware
//
#include <hip/hip_runtime.h>

// -------------------------------------------------------------------------
// B=128, N=64, CL=64, OUT=2*CL=128
// Pipeline:
//  kernel 1 (encode): h = relu(relu(s@se1)@se2); store hx,hy; P1/P2 = h@rc_w1
//    halves; Q1/Q2 = h@an_w1 halves (bf16); self_dyn -> sdout (f32)
//  kernel 2 (pair):  per (b, i-tile16): L1 = relu(P1[i]+P2[j]+d*w1l+b1) via
//    rank-1 combine, then bf16 WMMA GEMMs for rel L2/L3 and att L2,
//    sigmoid*relu combine, deterministic reduce over j and i-tile
//  kernel 3 (final): out = (sum_i sd + sum_rel/64)/64
// -------------------------------------------------------------------------

typedef __bf16 bf16_t;
typedef __attribute__((ext_vector_type(16))) __bf16 v16bf;
typedef __attribute__((ext_vector_type(8)))  __bf16 v8bf;
typedef __attribute__((ext_vector_type(8)))  float  v8f;

__device__ __forceinline__ v16bf cat8(v8bf lo, v8bf hi) {
  return __builtin_shufflevector(lo, hi, 0,1,2,3,4,5,6,7,8,9,10,11,12,13,14,15);
}

// A-fragment (16x32 bf16, this code's contiguous k order, same as B): lane l holds row (l&15);
// K = kstep*32 + (l>>4)*16 + {0..15}  -> two 16B LDS loads
__device__ __forceinline__ v16bf load_a_frag(const bf16_t* st, int stride,
                                             int kst, int ln, int hb) {
  const bf16_t* p = st + ln * stride + kst * 32 + hb * 16;
  v8bf lo = *(const v8bf*)p;
  v8bf hi = *(const v8bf*)(p + 8);
  return cat8(lo, hi);
}

// B-fragment staged fragment-linear in LDS: frag f, lane l -> 16 contiguous bf16
__device__ __forceinline__ v16bf load_b_frag(const bf16_t* base, int f, int l) {
  const bf16_t* p = base + (f * 32 + l) * 16;
  v8bf lo = *(const v8bf*)p;
  v8bf hi = *(const v8bf*)(p + 8);
  return cat8(lo, hi);
}

__device__ __forceinline__ v8f wmma_bf16(v16bf a, v16bf b, v8f c) {
  return __builtin_amdgcn_wmma_f32_16x16x32_bf16(false, a, false, b,
                                                 (short)0, c, false, false);
}

__device__ __forceinline__ v8f splat8(float v) {
  v8f c;
#pragma unroll
  for (int r = 0; r < 8; ++r) c[r] = v;
  return c;
}

// same-wave cross-lane LDS handoff: drain DS pipe (CDNA5 split counters)
__device__ __forceinline__ void wave_lds_sync() {
  __builtin_amdgcn_wave_barrier();
  asm volatile("s_wait_dscnt 0" ::: "memory");
}

// =========================================================================
// Kernel 1: encoder + self-dyn + P/Q precompute. 32 rows per block.
// =========================================================================
__global__ __launch_bounds__(256) void gh_encode_kernel(
    const float* __restrict__ s,
    const float* __restrict__ se_w1, const float* __restrict__ se_b1,
    const float* __restrict__ se_w2, const float* __restrict__ se_b2,
    const float* __restrict__ sd_w1, const float* __restrict__ sd_b1,
    const float* __restrict__ sd_w2, const float* __restrict__ sd_b2,
    const float* __restrict__ rc_w1, const float* __restrict__ an_w1,
    float* __restrict__ hx, float* __restrict__ hy,
    bf16_t* __restrict__ P1g, bf16_t* __restrict__ P2g,
    bf16_t* __restrict__ Q1g, bf16_t* __restrict__ Q2g,
    float* __restrict__ sdout) {
  __shared__ __align__(16) bf16_t W1b[64 * 64];
  __shared__ __align__(16) bf16_t W2b[64 * 64];
  __shared__ __align__(16) bf16_t RC1[128 * 128];   // rc_w1 rows 0..127
  __shared__ __align__(16) bf16_t AN1[128 * 64];    // an_w1 rows 0..127
  __shared__ __align__(16) bf16_t SD1[64 * 128];
  __shared__ __align__(16) bf16_t SD2[128 * 128];
  __shared__ float B1s[64], B2s[64], SB1[128], SB2[128];
  __shared__ __align__(16) float bufA[32 * 64];     // x -> h
  __shared__ __align__(16) float bufB[32 * 128];    // t -> u

  int tid = threadIdx.x;
  for (int i = tid; i < 64 * 64; i += 256) {
    W1b[i] = (bf16_t)se_w1[i];
    W2b[i] = (bf16_t)se_w2[i];
  }
  for (int i = tid; i < 128 * 128; i += 256) {
    RC1[i] = (bf16_t)rc_w1[i];
    SD2[i] = (bf16_t)sd_w2[i];
  }
  for (int i = tid; i < 128 * 64; i += 256) {
    AN1[i] = (bf16_t)an_w1[i];
    SD1[i] = (bf16_t)sd_w1[i];
  }
  if (tid < 64) { B1s[tid] = se_b1[tid]; B2s[tid] = se_b2[tid]; }
  if (tid < 128) { SB1[tid] = sd_b1[tid]; SB2[tid] = sd_b2[tid]; }
  int r0 = blockIdx.x * 32;
  for (int i = tid; i < 32 * 64; i += 256) bufA[i] = s[r0 * 64 + i];
  __syncthreads();

  // se layer 1 -> bufB (cols 0..63)
  for (int e = tid; e < 32 * 64; e += 256) {
    int row = e >> 6, c = e & 63;
    float acc = B1s[c];
    for (int k = 0; k < 64; ++k) acc += bufA[row * 64 + k] * (float)W1b[k * 64 + c];
    bufB[row * 128 + c] = fmaxf(acc, 0.f);
  }
  __syncthreads();
  // se layer 2 -> bufA (= h)
  for (int e = tid; e < 32 * 64; e += 256) {
    int row = e >> 6, c = e & 63;
    float acc = B2s[c];
    for (int k = 0; k < 64; ++k) acc += bufB[row * 128 + k] * (float)W2b[k * 64 + c];
    bufA[e] = fmaxf(acc, 0.f);
  }
  __syncthreads();
  if (tid < 32) {
    hx[r0 + tid] = bufA[tid * 64 + 0];
    hy[r0 + tid] = bufA[tid * 64 + 1];
  }
  // P1/P2 = h @ rc_w1[0:64], h @ rc_w1[64:128]
  for (int e = tid; e < 32 * 128; e += 256) {
    int row = e >> 7, c = e & 127;
    float a1 = 0.f, a2 = 0.f;
    for (int k = 0; k < 64; ++k) {
      float hv = bufA[row * 64 + k];
      a1 += hv * (float)RC1[k * 128 + c];
      a2 += hv * (float)RC1[(64 + k) * 128 + c];
    }
    P1g[(r0 + row) * 128 + c] = (bf16_t)a1;
    P2g[(r0 + row) * 128 + c] = (bf16_t)a2;
  }
  // Q1/Q2 = h @ an_w1 halves
  for (int e = tid; e < 32 * 64; e += 256) {
    int row = e >> 6, c = e & 63;
    float a1 = 0.f, a2 = 0.f;
    for (int k = 0; k < 64; ++k) {
      float hv = bufA[row * 64 + k];
      a1 += hv * (float)AN1[k * 64 + c];
      a2 += hv * (float)AN1[(64 + k) * 64 + c];
    }
    Q1g[(r0 + row) * 64 + c] = (bf16_t)a1;
    Q2g[(r0 + row) * 64 + c] = (bf16_t)a2;
  }
  // self-dyn layer 1 -> bufB
  for (int e = tid; e < 32 * 128; e += 256) {
    int row = e >> 7, c = e & 127;
    float acc = SB1[c];
    for (int k = 0; k < 64; ++k) acc += bufA[row * 64 + k] * (float)SD1[k * 128 + c];
    bufB[e] = fmaxf(acc, 0.f);
  }
  __syncthreads();
  // self-dyn layer 2 -> sdout (no relu)
  for (int e = tid; e < 32 * 128; e += 256) {
    int row = e >> 7, c = e & 127;
    float acc = SB2[c];
    for (int k = 0; k < 128; ++k) acc += bufB[row * 128 + k] * (float)SD2[k * 128 + c];
    sdout[(r0 + row) * 128 + c] = acc;
  }
}

// =========================================================================
// Kernel 2: fused all-pairs relation + attention via bf16 WMMA.
// One block per (b, 16-row i-tile); 8 waves, each owns 2 i's.
// =========================================================================
__global__ __launch_bounds__(256) void gh_pair_kernel(
    const float* __restrict__ rc_w1, const float* __restrict__ rc_b1,
    const float* __restrict__ rc_w2, const float* __restrict__ rc_b2,
    const float* __restrict__ rc_w3, const float* __restrict__ rc_b3,
    const float* __restrict__ an_w1, const float* __restrict__ an_b1,
    const float* __restrict__ an_w2, const float* __restrict__ an_b2,
    const bf16_t* __restrict__ P1g, const bf16_t* __restrict__ P2g,
    const bf16_t* __restrict__ Q1g, const bf16_t* __restrict__ Q2g,
    const float* __restrict__ hx, const float* __restrict__ hy,
    float* __restrict__ relsum) {
  __shared__ __align__(16) bf16_t fragRC2[32 * 32 * 16];  // 32 frags
  __shared__ __align__(16) bf16_t fragRC3[32 * 32 * 16];
  __shared__ __align__(16) bf16_t fragAN2[16 * 32 * 16];  // 16 frags
  __shared__ __align__(16) bf16_t stage[8][16 * 128];     // per-wave tile
  __shared__ __align__(16) bf16_t P1s[16 * 128];
  __shared__ __align__(16) bf16_t P2s[64 * 128];
  __shared__ __align__(16) bf16_t Q1s[16 * 64];
  __shared__ __align__(16) bf16_t Q2s[64 * 64];
  __shared__ float hxs[64], hys[64];
  __shared__ float w1lRC[128], b1RC[128], b2RC[128], b3RC[128];
  __shared__ float w1lAN[64], b1AN[64], b2AN[128];
  __shared__ float relred[16][128];

  int tid = threadIdx.x;
  int b = blockIdx.x >> 2;
  int it = blockIdx.x & 3;
  int i0 = it * 16;

  // weights -> fragment-linear LDS layout (B-matrix 32x16: lane half = K-half)
  for (int idx = tid; idx < 32 * 32 * 16; idx += 256) {
    int f = idx >> 9;
    int l = (idx >> 4) & 31;
    int t = idx & 15;
    int kst = f >> 3, nt = f & 7;
    int k = kst * 32 + (l >> 4) * 16 + t;
    int n = nt * 16 + (l & 15);
    fragRC2[idx] = (bf16_t)rc_w2[k * 128 + n];
    fragRC3[idx] = (bf16_t)rc_w3[k * 128 + n];
  }
  for (int idx = tid; idx < 16 * 32 * 16; idx += 256) {
    int f = idx >> 9;
    int l = (idx >> 4) & 31;
    int t = idx & 15;
    int kst = f >> 3, nt = f & 7;
    int k = kst * 32 + (l >> 4) * 16 + t;
    int n = nt * 16 + (l & 15);
    fragAN2[idx] = (bf16_t)an_w2[k * 128 + n];
  }
  for (int idx = tid; idx < 64 * 128; idx += 256) P2s[idx] = P2g[b * 64 * 128 + idx];
  for (int idx = tid; idx < 16 * 128; idx += 256) P1s[idx] = P1g[(b * 64 + i0) * 128 + idx];
  for (int idx = tid; idx < 64 * 64; idx += 256) Q2s[idx] = Q2g[b * 64 * 64 + idx];
  for (int idx = tid; idx < 16 * 64; idx += 256) Q1s[idx] = Q1g[(b * 64 + i0) * 64 + idx];
  if (tid < 64) {
    hxs[tid] = hx[b * 64 + tid];
    hys[tid] = hy[b * 64 + tid];
    w1lAN[tid] = an_w1[128 * 64 + tid];  // last row of an_w1 (dist weight)
    b1AN[tid] = an_b1[tid];
  }
  if (tid < 128) {
    w1lRC[tid] = rc_w1[128 * 128 + tid];  // last row of rc_w1
    b1RC[tid] = rc_b1[tid];
    b2RC[tid] = rc_b2[tid];
    b3RC[tid] = rc_b3[tid];
    b2AN[tid] = an_b2[tid];
  }
  __syncthreads();

  int w = tid >> 5;
  int l = tid & 31;
  int ln = l & 15;
  int hb = l >> 4;
  bf16_t* st = stage[w];

  for (int ii = 0; ii < 2; ++ii) {
    int il = w * 2 + ii;   // local i owned exclusively by this wave
    int ig = i0 + il;
    float racc[8];
#pragma unroll
    for (int nt = 0; nt < 8; ++nt) racc[nt] = 0.f;

    for (int jc = 0; jc < 4; ++jc) {
      int j0 = jc * 16;
      // ---- att layer 1 (rank-1 combine): A1[16x64] into stage ----
      {
        int row = l >> 1;
        int jg = j0 + row;
        float dx = hxs[ig] - hxs[jg], dy = hys[ig] - hys[jg];
        float d = dx * dx + dy * dy;
        int kb = (l & 1) * 32;
        for (int k2 = 0; k2 < 32; ++k2) {
          int k = kb + k2;
          float v = (float)Q1s[il * 64 + k] + (float)Q2s[jg * 64 + k] +
                    d * w1lAN[k] + b1AN[k];
          st[row * 64 + k] = (bf16_t)fmaxf(v, 0.f);
        }
      }
      wave_lds_sync();
      // ---- att layer 2: WMMA, bias folded into accumulator ----
      v8f att[8];
#pragma unroll
      for (int nt = 0; nt < 8; ++nt) att[nt] = splat8(b2AN[nt * 16 + ln]);
#pragma unroll
      for (int kst = 0; kst < 2; ++kst) {
        v16bf a = load_a_frag(st, 64, kst, ln, hb);
#pragma unroll
        for (int nt = 0; nt < 8; ++nt)
          att[nt] = wmma_bf16(a, load_b_frag(fragAN2, kst * 8 + nt, l), att[nt]);
      }
#pragma unroll
      for (int nt = 0; nt < 8; ++nt)
#pragma unroll
        for (int r2 = 0; r2 < 8; ++r2)
          att[nt][r2] = 1.f / (1.f + __expf(-att[nt][r2]));
      wave_lds_sync();
      // ---- rel layer 1 (rank-1 combine): L1[16x128] into stage ----
      {
        int row = l >> 1;
        int jg = j0 + row;
        float dx = hxs[ig] - hxs[jg], dy = hys[ig] - hys[jg];
        float d = dx * dx + dy * dy;
        int kb = (l & 1) * 64;
        for (int k2 = 0; k2 < 64; ++k2) {
          int k = kb + k2;
          float v = (float)P1s[il * 128 + k] + (float)P2s[jg * 128 + k] +
                    d * w1lRC[k] + b1RC[k];
          st[row * 128 + k] = (bf16_t)fmaxf(v, 0.f);
        }
      }
      wave_lds_sync();
      // ---- rel layer 2: WMMA ----
      v8f acc[8];
#pragma unroll
      for (int nt = 0; nt < 8; ++nt) acc[nt] = splat8(b2RC[nt * 16 + ln]);
#pragma unroll
      for (int kst = 0; kst < 4; ++kst) {
        v16bf a = load_a_frag(st, 128, kst, ln, hb);
#pragma unroll
        for (int nt = 0; nt < 8; ++nt)
          acc[nt] = wmma_bf16(a, load_b_frag(fragRC2, kst * 8 + nt, l), acc[nt]);
      }
      wave_lds_sync();
      // relu -> stage (C layout: lane holds rows hb*8+r at col nt*16+ln)
#pragma unroll
      for (int nt = 0; nt < 8; ++nt)
#pragma unroll
        for (int r2 = 0; r2 < 8; ++r2) {
          int row = hb * 8 + r2, col = nt * 16 + ln;
          st[row * 128 + col] = (bf16_t)fmaxf(acc[nt][r2], 0.f);
        }
      wave_lds_sync();
      // ---- rel layer 3: WMMA ----
#pragma unroll
      for (int nt = 0; nt < 8; ++nt) acc[nt] = splat8(b3RC[nt * 16 + ln]);
#pragma unroll
      for (int kst = 0; kst < 4; ++kst) {
        v16bf a = load_a_frag(st, 128, kst, ln, hb);
#pragma unroll
        for (int nt = 0; nt < 8; ++nt)
          acc[nt] = wmma_bf16(a, load_b_frag(fragRC3, kst * 8 + nt, l), acc[nt]);
      }
      // ---- combine: sum_j relu(L3) * sigmoid(att) (C layouts aligned) ----
#pragma unroll
      for (int nt = 0; nt < 8; ++nt)
#pragma unroll
        for (int r2 = 0; r2 < 8; ++r2)
          racc[nt] += fmaxf(acc[nt][r2], 0.f) * att[nt][r2];
      wave_lds_sync();
    }
    // fold the two lane halves (same column, different j rows)
#pragma unroll
    for (int nt = 0; nt < 8; ++nt) racc[nt] += __shfl_xor(racc[nt], 16, 32);
    if (hb == 0) {
#pragma unroll
      for (int nt = 0; nt < 8; ++nt) relred[il][nt * 16 + ln] = racc[nt];
    }
  }
  __syncthreads();
  if (tid < 128) {
    float ssum = 0.f;
#pragma unroll
    for (int i2 = 0; i2 < 16; ++i2) ssum += relred[i2][tid];
    relsum[(b * 4 + it) * 128 + tid] = ssum;
  }
}

// =========================================================================
// Kernel 3: out[b][c] = (sum_i sd[b,i,c] + sum_rel[b,c]/64) / 64
// =========================================================================
__global__ __launch_bounds__(256) void gh_final_kernel(
    const float* __restrict__ sdout, const float* __restrict__ relsum,
    float* __restrict__ out) {
  int idx = blockIdx.x * 256 + threadIdx.x;
  if (idx >= 128 * 128) return;
  int b = idx >> 7, c = idx & 127;
  float ss = 0.f;
  for (int i = 0; i < 64; ++i) ss += sdout[(b * 64 + i) * 128 + c];
  float rs = 0.f;
  for (int t = 0; t < 4; ++t) rs += relsum[(b * 4 + t) * 128 + c];
  out[idx] = (ss + rs * (1.0f / 64.f)) * (1.0f / 64.f);
}

// =========================================================================
extern "C" void kernel_launch(void* const* d_in, const int* in_sizes, int n_in,
                              void* d_out, int out_size, void* d_ws,
                              size_t ws_size, hipStream_t stream) {
  const float* s     = (const float*)d_in[0];
  const float* se_w1 = (const float*)d_in[1];
  const float* se_b1 = (const float*)d_in[2];
  const float* se_w2 = (const float*)d_in[3];
  const float* se_b2 = (const float*)d_in[4];
  const float* sd_w1 = (const float*)d_in[5];
  const float* sd_b1 = (const float*)d_in[6];
  const float* sd_w2 = (const float*)d_in[7];
  const float* sd_b2 = (const float*)d_in[8];
  const float* rc_w1 = (const float*)d_in[9];
  const float* rc_b1 = (const float*)d_in[10];
  const float* rc_w2 = (const float*)d_in[11];
  const float* rc_b2 = (const float*)d_in[12];
  const float* rc_w3 = (const float*)d_in[13];
  const float* rc_b3 = (const float*)d_in[14];
  const float* an_w1 = (const float*)d_in[15];
  const float* an_b1 = (const float*)d_in[16];
  const float* an_w2 = (const float*)d_in[17];
  const float* an_b2 = (const float*)d_in[18];

  char* ws = (char*)d_ws;
  float*  hx     = (float*)(ws + 0);          //  32768 B
  float*  hy     = (float*)(ws + 32768);      //  32768 B
  bf16_t* P1g    = (bf16_t*)(ws + 65536);     // 2097152 B
  bf16_t* P2g    = (bf16_t*)(ws + 2162688);   // 2097152 B
  bf16_t* Q1g    = (bf16_t*)(ws + 4259840);   // 1048576 B
  bf16_t* Q2g    = (bf16_t*)(ws + 5308416);   // 1048576 B
  float*  sdout  = (float*)(ws + 6356992);    // 4194304 B
  float*  relsum = (float*)(ws + 10551296);   //  262144 B  (end 10813440)

  gh_encode_kernel<<<256, 256, 0, stream>>>(
      s, se_w1, se_b1, se_w2, se_b2, sd_w1, sd_b1, sd_w2, sd_b2, rc_w1, an_w1,
      hx, hy, P1g, P2g, Q1g, Q2g, sdout);

  gh_pair_kernel<<<512, 256, 0, stream>>>(
      rc_w1, rc_b1, rc_w2, rc_b2, rc_w3, rc_b3, an_w1, an_b1, an_w2, an_b2,
      P1g, P2g, Q1g, Q2g, hx, hy, relsum);

  gh_final_kernel<<<64, 256, 0, stream>>>(sdout, relsum, (float*)d_out);
}
